// BinarizedConv2d_19164144075029
// MI455X (gfx1250) — compile-verified, matrix operand order checked by reading, not run on hardware
//
#include <hip/hip_runtime.h>

// ---------------- CDNA5 WMMA implicit-GEMM binarized conv ----------------
// D = sign(W)[256 x 2304] x im2col(X)[2304 x 93312], f16 inputs, f32 accum.
// K reordered as k' = (r*3+s)*256 + c so each 32-wide K tile has fixed (r,s)
// and contiguous channels -> immediate-offset strided gathers, no div/mod.
// Requires ws_size >= 256*2304*2 = 1,179,648 bytes (pre-binarized weights).

typedef _Float16 f16;
typedef __attribute__((ext_vector_type(16))) _Float16 v16h;
typedef __attribute__((ext_vector_type(8)))  _Float16 v8h;
typedef __attribute__((ext_vector_type(8)))  float    v8f;

#define TILE_M 128
#define TILE_N 128
#define TILE_K 32
#define LDS_STRIDE 40        // padded K-stride in f16 elems: 80B, 16B aligned

#define C_IN    256
#define K_TOT   (C_IN * 9)   // 2304
#define NKT     (K_TOT / TILE_K)  // 72
#define H_IN    56
#define W_IN    56
#define CHW_IN  (C_IN * H_IN * W_IN)
#define HW_IN   (H_IN * W_IN)
#define H_OUT   54
#define W_OUT   54
#define SPATIAL (H_OUT * W_OUT)   // 2916
#define N_IMG   32
#define N_TOT   (N_IMG * SPATIAL) // 93312 = 729 * 128

// ---- one-shot: binarize W and transpose K to k' = rs*256 + c (f16) ----
__global__ __launch_bounds__(256)
void pack_w_sign(const float* __restrict__ W, f16* __restrict__ A)
{
    const int t  = blockIdx.x * 256 + threadIdx.x;   // dst index, coalesced
    const int m  = t / K_TOT;
    const int q  = t - m * K_TOT;
    const int rs = q >> 8;           // 0..8
    const int c  = q & 255;
    const float w = W[m * K_TOT + c * 9 + rs];
    A[t] = (w >= 0.f) ? (f16)1.0f : (f16)-1.0f;
}

__global__ __launch_bounds__(256)
void binconv_wmma_f16(const float* __restrict__ X,
                      const f16*  __restrict__ A,    // [256][2304], k'-ordered
                      float* __restrict__ out)
{
    __shared__ f16 sA[TILE_M * LDS_STRIDE];   // 10240 B
    __shared__ f16 sB[TILE_N * LDS_STRIDE];   // 10240 B

    const int tid   = threadIdx.x;
    const int lane  = tid & 31;
    const int wv    = tid >> 5;
    const int waveM = wv & 1;          // 2 waves along M (64 rows each)
    const int waveN = wv >> 1;         // 4 waves along N (32 cols each)
    const int lmod  = lane & 15;
    const int lhalf = lane >> 4;

    const int nBase  = blockIdx.x * TILE_N;   // 729 tiles
    const int ocBase = blockIdx.y * TILE_M;   // 2 tiles

    // ---- A loader: 2 threads per M-row, 16 contiguous f16 each ----
    const int aRow  = tid >> 1;               // 0..127
    const int aKoff = (tid & 1) * 16;         // 0 or 16
    const f16* aPtr = A + (size_t)(ocBase + aRow) * K_TOT + aKoff;

    // ---- B loader: one n per thread (fixed), 16 consecutive channels ----
    const int nLoc  = tid & 127;
    const int kHalf = tid >> 7;               // 0 or 1
    const int nGlob = nBase + nLoc;
    const int imgL  = nGlob / SPATIAL;
    const int pL    = nGlob - imgL * SPATIAL;
    const int ohL   = pL / W_OUT;
    const int owL   = pL - ohL * W_OUT;
    const float* xBase = X + (size_t)imgL * CHW_IN + ohL * W_IN + owL;

    v8f acc[4][2] = {};

    // ---- prologue: prefetch tile kt=0 into registers ----
    uint4 aReg0 = *(const uint4*)(aPtr);          // k' 0..7
    uint4 aReg1 = *(const uint4*)(aPtr + 8);      // k' 8..15
    float xv[16];
    {
        const float* xp = xBase + (size_t)(kHalf * 16) * HW_IN;  // rs=0
        #pragma unroll
        for (int i = 0; i < 16; ++i) xv[i] = xp[i * HW_IN];
    }

    for (int kt = 0; kt < NKT; ++kt) {
        __syncthreads();   // previous tile's fragment reads complete

        // ---- fill LDS from prefetched registers ----
        *(uint4*)&sA[aRow * LDS_STRIDE + aKoff]     = aReg0;
        *(uint4*)&sA[aRow * LDS_STRIDE + aKoff + 8] = aReg1;
        #pragma unroll
        for (int i = 0; i < 8; ++i) {
            const unsigned packed = __builtin_bit_cast(
                unsigned, __builtin_amdgcn_cvt_pkrtz(xv[2*i], xv[2*i+1]));
            *(unsigned*)&sB[nLoc * LDS_STRIDE + kHalf * 16 + 2*i] = packed;
        }

        // ---- prefetch next tile (overlaps ds loads + WMMA below) ----
        if (kt + 1 < NKT) {
            const int ktn = kt + 1;
            const int k0n = ktn * TILE_K;
            aReg0 = *(const uint4*)(aPtr + k0n);
            aReg1 = *(const uint4*)(aPtr + k0n + 8);
            const int rsn = ktn >> 3;              // uniform: filter tap
            const int rn  = rsn / 3;
            const int sn  = rsn - rn * 3;
            const int c0n = (ktn & 7) * TILE_K;    // uniform: channel base
            const float* xp = xBase + rn * W_IN + sn
                            + (size_t)(c0n + kHalf * 16) * HW_IN;
            #pragma unroll
            for (int i = 0; i < 16; ++i) xv[i] = xp[i * HW_IN];
        }

        __syncthreads();   // LDS tile visible

        // ---- fragments (ISA 7.12.2: lanes 0-15 K 0-7 & 16-23, etc.) ----
        v16h afrag[4];
        #pragma unroll
        for (int i = 0; i < 4; ++i) {
            const int row = waveM * 64 + i * 16 + lmod;
            const f16* base = &sA[row * LDS_STRIDE + lhalf * 8];
            const v8h lo = *(const v8h*)(base);
            const v8h hi = *(const v8h*)(base + 16);
            afrag[i] = __builtin_shufflevector(lo, hi,
                0,1,2,3,4,5,6,7,8,9,10,11,12,13,14,15);
        }
        v16h bfrag[2];
        #pragma unroll
        for (int j = 0; j < 2; ++j) {
            const int row = waveN * 32 + j * 16 + lmod;    // sB is N-major
            const f16* base = &sB[row * LDS_STRIDE + lhalf * 8];
            const v8h lo = *(const v8h*)(base);
            const v8h hi = *(const v8h*)(base + 16);
            bfrag[j] = __builtin_shufflevector(lo, hi,
                0,1,2,3,4,5,6,7,8,9,10,11,12,13,14,15);
        }

        // ---- 8 x V_WMMA_F32_16X16X32_F16 ----
        #pragma unroll
        for (int i = 0; i < 4; ++i)
            #pragma unroll
            for (int j = 0; j < 2; ++j)
                acc[i][j] = __builtin_amdgcn_wmma_f32_16x16x32_f16(
                    false, afrag[i], false, bfrag[j],
                    (short)0, acc[i][j], false, false);
    }

    // ---- store C (lanes 0-15: M=v, lanes 16-31: M=v+8) ----
    #pragma unroll
    for (int j = 0; j < 2; ++j) {
        const int n   = nBase + waveN * 32 + j * 16 + lmod;
        const int img = n / SPATIAL;
        const int p   = n - img * SPATIAL;
        float* obase  = out + (size_t)img * (256 * SPATIAL) + p;
        #pragma unroll
        for (int i = 0; i < 4; ++i) {
            const int ocRow = ocBase + waveM * 64 + i * 16 + lhalf * 8;
            #pragma unroll
            for (int v = 0; v < 8; ++v)
                obase[(size_t)(ocRow + v) * SPATIAL] = acc[i][j][v];
        }
    }
}

extern "C" void kernel_launch(void* const* d_in, const int* in_sizes, int n_in,
                              void* d_out, int out_size, void* d_ws, size_t ws_size,
                              hipStream_t stream) {
    (void)in_sizes; (void)n_in; (void)ws_size; (void)out_size;
    const float* X = (const float*)d_in[0];   // [32,256,56,56] f32
    const float* W = (const float*)d_in[1];   // [256,256,3,3]  f32
    float* out = (float*)d_out;               // [32,256,54,54] f32
    f16* Apack = (f16*)d_ws;                  // 1.18 MB scratch

    // one-shot weight binarize + K-reorder (stream-ordered before main GEMM)
    pack_w_sign<<<dim3((256 * K_TOT) / 256), dim3(256), 0, stream>>>(W, Apack);

    dim3 grid(N_TOT / TILE_N, 256 / TILE_M);  // (729, 2)
    dim3 block(256);                          // 8 wave32
    binconv_wmma_f16<<<grid, block, 0, stream>>>(X, Apack, out);
}
